// MambaBlock_48859547959377
// MI455X (gfx1250) — matrix operand order repaired and checked by reading, not compiled, not run on hardware
//
#include <hip/hip_runtime.h>
#include <hip/hip_bf16.h>

// Problem constants (match reference)
#define NB    2        // batch
#define LB    2048     // sequence length
#define DIN   512
#define DM    1024
#define NS    16       // ssm state
#define DR    64       // dt rank
#define KCONV 4
#define MTOT  (NB * LB)   // 4096 rows in all GEMMs
#define NSEG  16          // scan segments
#define TSEG  (LB / NSEG) // 128 timesteps per segment

typedef __attribute__((ext_vector_type(8)))  _Float16 v8h;
typedef __attribute__((ext_vector_type(16))) _Float16 v16h;
typedef __attribute__((ext_vector_type(8)))  float    v8f;

// ---------------------------------------------------------------------------
// WMMA helpers (gfx1250, wave32). Layouts used by this code (A/B K order is contiguous, same for both):
//  A 16x32 f16 : lane -> (M = lane&15, Kblk = lane>>4); elems 0..15 = 16 contiguous K values
//  B 32x16 f16 : from row-major (N,K) weight: lane -> (N = lane&15, Kblk = lane>>4),
//                elems 0..15 = 16 contiguous K values
//  C/D 16x16 f32: elem r -> row r + 8*(lane>>4), col = lane&15
// ---------------------------------------------------------------------------
__device__ __forceinline__ v16h load_a_frag(const _Float16* __restrict__ A, int lda,
                                            int m0, int k0) {
  const int lane = threadIdx.x & 31;
  const int m  = m0 + (lane & 15);
  const int kb = (lane >> 4) << 4;                  // 0 or 16
  const _Float16* p = A + (size_t)m * lda + k0 + kb;
  v8h lo = *(const v8h*)p;
  v8h hi = *(const v8h*)(p + 8);
  v16h r;
#pragma unroll
  for (int i = 0; i < 8; ++i) { r[i] = lo[i]; r[i + 8] = hi[i]; }
  return r;
}

__device__ __forceinline__ v16h load_b_frag(const _Float16* __restrict__ W, int ldw,
                                            int n0, int k0) {
  const int lane = threadIdx.x & 31;
  const int n  = n0 + (lane & 15);
  const int kb = (lane >> 4) << 4;                  // 0 or 16
  return *(const v16h*)(W + (size_t)n * ldw + k0 + kb);
}

__device__ __forceinline__ v8f wmma16(v16h a, v16h b, v8f c) {
  return __builtin_amdgcn_wmma_f32_16x16x32_f16(false, a, false, b, (short)0, c,
                                                false, false);
}

__device__ __forceinline__ void store_c_f32(float* __restrict__ C, int ldc,
                                            int m0, int n0, v8f c) {
  const int lane = threadIdx.x & 31;
  const int col = lane & 15;
  const int rb  = (lane >> 4) << 3;
#pragma unroll
  for (int r = 0; r < 8; ++r)
    C[(size_t)(m0 + rb + r) * ldc + n0 + col] = c[r];
}

__device__ __forceinline__ float silu(float v) { return v / (1.f + __expf(-v)); }

// ---------------------------------------------------------------------------
// fp32 -> fp16 conversion (weights + activations staging)
// ---------------------------------------------------------------------------
__global__ void k_cvt_f16(const float* __restrict__ s, _Float16* __restrict__ d, int n) {
  const int i = blockIdx.x * 256 + threadIdx.x;
  if (i < n) d[i] = (_Float16)s[i];
}

// ---------------------------------------------------------------------------
// K1: x_pre = x_in @ W_in^T ; res_act = silu(x_in @ W_res^T)
//     M=4096, N=1024, K=512. Register-blocked: each wave does a 32x32 output
//     block for BOTH weight matrices: 2 A-frags + 4 B-frags -> 8 WMMAs/k-step.
// ---------------------------------------------------------------------------
__global__ void k_gemm_in(const _Float16* __restrict__ A,
                          const _Float16* __restrict__ W0,
                          const _Float16* __restrict__ W1,
                          float* __restrict__ X, float* __restrict__ RA) {
  const int w  = (blockIdx.x << 3) + (threadIdx.x >> 5);
  const int m0 = (w >> 5) << 5;                     // 128 M-blocks of 32
  const int n0 = (w & 31) << 5;                     // 32 N-blocks of 32
  v8f c0[4] = {{}, {}, {}, {}};                     // X accum [mi*2+ni]
  v8f c1[4] = {{}, {}, {}, {}};                     // res accum
  for (int k0 = 0; k0 < DIN; k0 += 32) {
    v16h a0 = load_a_frag(A, DIN, m0, k0);
    v16h a1 = load_a_frag(A, DIN, m0 + 16, k0);
    v16h p0 = load_b_frag(W0, DIN, n0, k0);
    v16h p1 = load_b_frag(W0, DIN, n0 + 16, k0);
    v16h q0 = load_b_frag(W1, DIN, n0, k0);
    v16h q1 = load_b_frag(W1, DIN, n0 + 16, k0);
    c0[0] = wmma16(a0, p0, c0[0]);  c0[1] = wmma16(a0, p1, c0[1]);
    c0[2] = wmma16(a1, p0, c0[2]);  c0[3] = wmma16(a1, p1, c0[3]);
    c1[0] = wmma16(a0, q0, c1[0]);  c1[1] = wmma16(a0, q1, c1[1]);
    c1[2] = wmma16(a1, q0, c1[2]);  c1[3] = wmma16(a1, q1, c1[3]);
  }
  const int lane = threadIdx.x & 31;
  const int col = lane & 15, rb = (lane >> 4) << 3;
#pragma unroll
  for (int mi = 0; mi < 2; ++mi)
#pragma unroll
    for (int ni = 0; ni < 2; ++ni) {
      v8f cx = c0[mi * 2 + ni], cr = c1[mi * 2 + ni];
#pragma unroll
      for (int r = 0; r < 8; ++r) {
        const size_t o =
            (size_t)(m0 + mi * 16 + rb + r) * DM + n0 + ni * 16 + col;
        X[o]  = cx[r];
        RA[o] = silu(cr[r]);                        // fused silu(res)
      }
    }
}

// ---------------------------------------------------------------------------
// K2: causal depthwise conv1d (K=4, left pad 3) + bias + silu
// ---------------------------------------------------------------------------
__global__ void k_conv_silu(const float* __restrict__ xp, const float* __restrict__ cw,
                            const float* __restrict__ cb, float* __restrict__ xs,
                            _Float16* __restrict__ xs16) {
  const int i  = blockIdx.x * 256 + threadIdx.x;    // over MTOT*DM
  const int d  = i & (DM - 1);
  const int bl = i >> 10;                           // b*L + t
  const int t  = bl & (LB - 1);
  float acc = cb[d];
#pragma unroll
  for (int j = 0; j < KCONV; ++j) {
    const int tt = t - (KCONV - 1) + j;
    if (tt >= 0) acc += xp[(size_t)(bl - t + tt) * DM + d] * cw[d * KCONV + j];
  }
  const float s = silu(acc);
  xs[i]   = s;
  xs16[i] = (_Float16)s;
}

// ---------------------------------------------------------------------------
// K3: Bp = xs@W_B^T (16), Cp = xs@W_C^T (16), dtr = xs@W_dt^T (64)
//     One M-tile per wave; A-fragment reused across all 6 N-tiles.
// ---------------------------------------------------------------------------
__global__ void k_gemm_bcdt(const _Float16* __restrict__ A,
                            const _Float16* __restrict__ WB,
                            const _Float16* __restrict__ WC,
                            const _Float16* __restrict__ WD,
                            float* __restrict__ Bp, float* __restrict__ Cp,
                            _Float16* __restrict__ dtr) {
  const int w  = (blockIdx.x << 3) + (threadIdx.x >> 5);
  const int m0 = w << 4;
  v8f cB = {}, cC = {};
  v8f cD[4] = {{}, {}, {}, {}};
  for (int k0 = 0; k0 < DM; k0 += 32) {
    v16h a = load_a_frag(A, DM, m0, k0);
    cB = wmma16(a, load_b_frag(WB, DM, 0, k0), cB);
    cC = wmma16(a, load_b_frag(WC, DM, 0, k0), cC);
#pragma unroll
    for (int j = 0; j < 4; ++j)
      cD[j] = wmma16(a, load_b_frag(WD, DM, j << 4, k0), cD[j]);
  }
  const int lane = threadIdx.x & 31;
  const int col = lane & 15, rb = (lane >> 4) << 3;
#pragma unroll
  for (int r = 0; r < 8; ++r) {
    const int m = m0 + rb + r;
    Bp[(size_t)m * NS + col] = cB[r];
    Cp[(size_t)m * NS + col] = cC[r];
#pragma unroll
    for (int j = 0; j < 4; ++j)
      dtr[(size_t)m * DR + (j << 4) + col] = (_Float16)cD[j][r];
  }
}

// ---------------------------------------------------------------------------
// K4: delta = softplus(dtr @ W_dtproj^T + b)  M=4096, N=1024, K=64
//     Register-blocked 32x64 per wave (2 A-frags x 4 B-frags).
// ---------------------------------------------------------------------------
__global__ void k_gemm_delta(const _Float16* __restrict__ A,
                             const _Float16* __restrict__ W,
                             const float* __restrict__ bias,
                             float* __restrict__ delta) {
  const int w  = (blockIdx.x << 3) + (threadIdx.x >> 5);
  const int m0 = (w >> 4) << 5;                     // 128 M-blocks of 32
  const int n0 = (w & 15) << 6;                     // 16 N-blocks of 64
  v8f c[8] = {{}, {}, {}, {}, {}, {}, {}, {}};      // [mi*4+ni]
  for (int k0 = 0; k0 < DR; k0 += 32) {
    v16h a0 = load_a_frag(A, DR, m0, k0);
    v16h a1 = load_a_frag(A, DR, m0 + 16, k0);
#pragma unroll
    for (int ni = 0; ni < 4; ++ni) {
      v16h b = load_b_frag(W, DR, n0 + ni * 16, k0);
      c[ni]     = wmma16(a0, b, c[ni]);
      c[4 + ni] = wmma16(a1, b, c[4 + ni]);
    }
  }
  const int lane = threadIdx.x & 31;
  const int col = lane & 15, rb = (lane >> 4) << 3;
#pragma unroll
  for (int mi = 0; mi < 2; ++mi)
#pragma unroll
    for (int ni = 0; ni < 4; ++ni) {
      const float bb = bias[n0 + ni * 16 + col];
      v8f cc = c[mi * 4 + ni];
#pragma unroll
      for (int r = 0; r < 8; ++r) {
        const float z = cc[r] + bb;
        delta[(size_t)(m0 + mi * 16 + rb + r) * DM + n0 + ni * 16 + col] =
            (z > 20.f) ? z : log1pf(__expf(z));
      }
    }
}

// ---------------------------------------------------------------------------
// K5: segmented selective scan (h_t = a_t h_{t-1} + b_t is associative).
//  Pass A: per (segment, channel): P = prod(dA), c = segment contribution.
//  Combine: chain boundary states across segments (16 cheap serial steps).
//  Pass B: replay each segment from its true start state, emit gated output.
// ---------------------------------------------------------------------------
__global__ void k_scan_seg(const float* __restrict__ delta, const float* __restrict__ xs,
                           const float* __restrict__ Bp, const float* __restrict__ A_log,
                           float* __restrict__ segP, float* __restrict__ segC) {
  __shared__ float sB[TSEG * NS];                   // 8 KB
  const int tid = threadIdx.x;
  const int d = ((blockIdx.x & 3) << 8) + tid;
  const int b = (blockIdx.x >> 2) & 1;
  const int s = blockIdx.x >> 3;
  float Arow[NS];
#pragma unroll
  for (int n = 0; n < NS; ++n) Arow[n] = -__expf(A_log[d * NS + n]);
  const size_t bc = ((size_t)b * LB + s * TSEG) * NS;
  for (int i = tid; i < TSEG * NS; i += 256) sB[i] = Bp[bc + i];
  __syncthreads();

  float P[NS], c[NS];
#pragma unroll
  for (int n = 0; n < NS; ++n) { P[n] = 1.f; c[n] = 0.f; }
  for (int tt = 0; tt < TSEG; ++tt) {
    const size_t idx = ((size_t)b * LB + s * TSEG + tt) * DM + d;
    const float dl = delta[idx];
    const float du = dl * xs[idx];
#pragma unroll
    for (int n = 0; n < NS; ++n) {
      const float dA = __expf(dl * Arow[n]);
      P[n] *= dA;
      c[n] = fmaf(dA, c[n], du * sB[tt * NS + n]);
    }
  }
  const size_t o = (((size_t)s * NB + b) * DM + d) * NS;
#pragma unroll
  for (int n = 0; n < NS; ++n) { segP[o + n] = P[n]; segC[o + n] = c[n]; }
}

__global__ void k_scan_combine(const float* __restrict__ segP,
                               const float* __restrict__ segC,
                               float* __restrict__ hstart) {
  const int id = blockIdx.x * 256 + threadIdx.x;    // over NB*DM*NS = 32768
  const size_t stride = (size_t)NB * DM * NS;       // per-segment stride
  float h = 0.f;
  for (int s = 0; s < NSEG; ++s) {
    const size_t o = (size_t)s * stride + id;
    hstart[o] = h;
    h = fmaf(segP[o], h, segC[o]);
  }
}

__global__ void k_scan_apply(const float* __restrict__ delta, const float* __restrict__ xs,
                             const float* __restrict__ resa, const float* __restrict__ Bp,
                             const float* __restrict__ Cp, const float* __restrict__ A_log,
                             const float* __restrict__ Dp, const float* __restrict__ hstart,
                             _Float16* __restrict__ y16) {
  __shared__ float sB[TSEG * NS];                   // 8 KB
  __shared__ float sC[TSEG * NS];                   // 8 KB
  const int tid = threadIdx.x;
  const int d = ((blockIdx.x & 3) << 8) + tid;
  const int b = (blockIdx.x >> 2) & 1;
  const int s = blockIdx.x >> 3;
  float Arow[NS];
#pragma unroll
  for (int n = 0; n < NS; ++n) Arow[n] = -__expf(A_log[d * NS + n]);
  const float dpar = Dp[d];
  const size_t bc = ((size_t)b * LB + s * TSEG) * NS;
  for (int i = tid; i < TSEG * NS; i += 256) {
    sB[i] = Bp[bc + i];
    sC[i] = Cp[bc + i];
  }
  __syncthreads();

  float h[NS];
  const size_t ho = (((size_t)s * NB + b) * DM + d) * NS;
#pragma unroll
  for (int n = 0; n < NS; ++n) h[n] = hstart[ho + n];
  for (int tt = 0; tt < TSEG; ++tt) {
    const size_t idx = ((size_t)b * LB + s * TSEG + tt) * DM + d;
    const float dl = delta[idx];
    const float xv = xs[idx];
    const float du = dl * xv;
    float y = 0.f;
#pragma unroll
    for (int n = 0; n < NS; ++n) {
      const float dA = __expf(dl * Arow[n]);
      h[n] = fmaf(dA, h[n], du * sB[tt * NS + n]);
      y = fmaf(h[n], sC[tt * NS + n], y);
    }
    y16[idx] = (_Float16)((y + xv * dpar) * resa[idx]);  // resa = silu(res)
  }
}

// ---------------------------------------------------------------------------
// K6: out = y @ W_out^T   M=4096, N=512, K=1024. Blocked 32x64 per wave.
// ---------------------------------------------------------------------------
__global__ void k_gemm_out(const _Float16* __restrict__ A,
                           const _Float16* __restrict__ W,
                           float* __restrict__ out) {
  const int w  = (blockIdx.x << 3) + (threadIdx.x >> 5);
  const int m0 = (w >> 3) << 5;                     // 128 M-blocks of 32
  const int n0 = (w & 7) << 6;                      // 8 N-blocks of 64
  v8f c[8] = {{}, {}, {}, {}, {}, {}, {}, {}};
  for (int k0 = 0; k0 < DM; k0 += 32) {
    v16h a0 = load_a_frag(A, DM, m0, k0);
    v16h a1 = load_a_frag(A, DM, m0 + 16, k0);
#pragma unroll
    for (int ni = 0; ni < 4; ++ni) {
      v16h b = load_b_frag(W, DM, n0 + ni * 16, k0);
      c[ni]     = wmma16(a0, b, c[ni]);
      c[4 + ni] = wmma16(a1, b, c[4 + ni]);
    }
  }
#pragma unroll
  for (int mi = 0; mi < 2; ++mi)
#pragma unroll
    for (int ni = 0; ni < 4; ++ni)
      store_c_f32(out, DIN, m0 + mi * 16, n0 + ni * 16, c[mi * 4 + ni]);
}

// ---------------------------------------------------------------------------
extern "C" void kernel_launch(void* const* d_in, const int* in_sizes, int n_in,
                              void* d_out, int out_size, void* d_ws, size_t ws_size,
                              hipStream_t stream) {
  (void)in_sizes; (void)n_in; (void)out_size; (void)ws_size;
  const float* x_in     = (const float*)d_in[0];
  const float* W_in     = (const float*)d_in[1];
  const float* W_res    = (const float*)d_in[2];
  const float* W_out    = (const float*)d_in[3];
  const float* conv_w   = (const float*)d_in[4];
  const float* conv_b   = (const float*)d_in[5];
  const float* A_log    = (const float*)d_in[6];
  const float* D_param  = (const float*)d_in[7];
  const float* W_B      = (const float*)d_in[8];
  const float* W_C      = (const float*)d_in[9];
  const float* W_dt     = (const float*)d_in[10];
  const float* W_dtproj = (const float*)d_in[11];
  const float* b_dtproj = (const float*)d_in[12];
  float* out = (float*)d_out;

  // Workspace carve-out (256B aligned), ~99 MB total
  char* ws = (char*)d_ws;
  size_t off = 0;
  auto carve = [&](size_t bytes) -> void* {
    void* p = ws + off;
    off = (off + bytes + 255) & ~(size_t)255;
    return p;
  };
  _Float16* x_in16     = (_Float16*)carve((size_t)MTOT * DIN * 2);
  _Float16* W_in16     = (_Float16*)carve((size_t)DM * DIN * 2);
  _Float16* W_res16    = (_Float16*)carve((size_t)DM * DIN * 2);
  _Float16* W_out16    = (_Float16*)carve((size_t)DIN * DM * 2);
  _Float16* W_B16      = (_Float16*)carve((size_t)NS * DM * 2);
  _Float16* W_C16      = (_Float16*)carve((size_t)NS * DM * 2);
  _Float16* W_dt16     = (_Float16*)carve((size_t)DR * DM * 2);
  _Float16* W_dtproj16 = (_Float16*)carve((size_t)DM * DR * 2);
  float*    x_pre      = (float*)carve((size_t)MTOT * DM * 4);
  float*    res_act    = (float*)carve((size_t)MTOT * DM * 4);
  float*    xs         = (float*)carve((size_t)MTOT * DM * 4);
  _Float16* xs16       = (_Float16*)carve((size_t)MTOT * DM * 2);
  float*    Bp         = (float*)carve((size_t)MTOT * NS * 4);
  float*    Cp         = (float*)carve((size_t)MTOT * NS * 4);
  _Float16* dtr16      = (_Float16*)carve((size_t)MTOT * DR * 2);
  float*    delta      = (float*)carve((size_t)MTOT * DM * 4);
  _Float16* y16        = (_Float16*)carve((size_t)MTOT * DM * 2);
  float*    segP       = (float*)carve((size_t)NSEG * NB * DM * NS * 4);
  float*    segC       = (float*)carve((size_t)NSEG * NB * DM * NS * 4);
  float*    hstart     = (float*)carve((size_t)NSEG * NB * DM * NS * 4);

  auto cdiv = [](int a, int b) { return (a + b - 1) / b; };

  // fp16 staging
  k_cvt_f16<<<cdiv(MTOT * DIN, 256), 256, 0, stream>>>(x_in, x_in16, MTOT * DIN);
  k_cvt_f16<<<cdiv(DM * DIN, 256), 256, 0, stream>>>(W_in, W_in16, DM * DIN);
  k_cvt_f16<<<cdiv(DM * DIN, 256), 256, 0, stream>>>(W_res, W_res16, DM * DIN);
  k_cvt_f16<<<cdiv(DIN * DM, 256), 256, 0, stream>>>(W_out, W_out16, DIN * DM);
  k_cvt_f16<<<cdiv(NS * DM, 256), 256, 0, stream>>>(W_B, W_B16, NS * DM);
  k_cvt_f16<<<cdiv(NS * DM, 256), 256, 0, stream>>>(W_C, W_C16, NS * DM);
  k_cvt_f16<<<cdiv(DR * DM, 256), 256, 0, stream>>>(W_dt, W_dt16, DR * DM);
  k_cvt_f16<<<cdiv(DM * DR, 256), 256, 0, stream>>>(W_dtproj, W_dtproj16, DM * DR);

  // K1: dual in-projection, 32x32 blocks per wave -> 4096 waves
  k_gemm_in<<<(MTOT / 32) * (DM / 32) / 8, 256, 0, stream>>>(
      x_in16, W_in16, W_res16, x_pre, res_act);

  // K2: causal depthwise conv + silu
  k_conv_silu<<<MTOT * DM / 256, 256, 0, stream>>>(x_pre, conv_w, conv_b, xs, xs16);

  // K3: B/C/dt_rank projections (256 M-tiles)
  k_gemm_bcdt<<<(MTOT / 16) / 8, 256, 0, stream>>>(
      xs16, W_B16, W_C16, W_dt16, Bp, Cp, dtr16);

  // K4: delta = softplus(dtr @ W_dtproj^T + b), 32x64 blocks per wave
  k_gemm_delta<<<(MTOT / 32) * (DM / 64) / 8, 256, 0, stream>>>(
      dtr16, W_dtproj16, b_dtproj, delta);

  // K5: segmented scan: per-segment transfer, boundary chain, replay
  k_scan_seg<<<NSEG * NB * (DM / 256), 256, 0, stream>>>(
      delta, xs, Bp, A_log, segP, segC);
  k_scan_combine<<<NB * DM * NS / 256, 256, 0, stream>>>(segP, segC, hstart);
  k_scan_apply<<<NSEG * NB * (DM / 256), 256, 0, stream>>>(
      delta, xs, res_act, Bp, Cp, A_log, D_param, hstart, y16);

  // K6: out-projection, 32x64 blocks per wave -> d_out
  k_gemm_out<<<(MTOT / 32) * (DIN / 64) / 8, 256, 0, stream>>>(y16, W_out16, out);
}
